// VectorQuantizer_4707284157218
// MI455X (gfx1250) — matrix operand order repaired and checked by reading, not compiled, not run on hardware
//
#include <hip/hip_runtime.h>
#include <hip/hip_bf16.h>

#define NUM_EMB 1024
#define EDIM    256
#define HW      4096         // 64*64
#define M_TOTAL 65536        // 16*4096
#define OUT_PLANE 16777216   // 16*256*4096

typedef __attribute__((ext_vector_type(16))) __bf16 v16bf;
typedef __attribute__((ext_vector_type(8)))  __bf16 v8bf;
typedef __attribute__((ext_vector_type(8)))  float  v8f;

// ---------------- kernel 1: bf16 codebook + norms + zero counts ----------------
__global__ void vq_prep(const float* __restrict__ emb, __bf16* __restrict__ ebf,
                        float* __restrict__ enorm, unsigned* __restrict__ counts) {
    const int n = blockIdx.x;      // code row
    const int t = threadIdx.x;     // element 0..255
    float v = emb[(size_t)n * EDIM + t];
    ebf[(size_t)n * EDIM + t] = (__bf16)v;
    float p = v * v;
#pragma unroll
    for (int off = 16; off; off >>= 1) p += __shfl_xor(p, off, 32);
    __shared__ float red[8];
    if ((t & 31) == 0) red[t >> 5] = p;
    __syncthreads();
    if (t == 0) {
        float s = 0.f;
#pragma unroll
        for (int i = 0; i < 8; ++i) s += red[i];
        enorm[n] = s;
        counts[n] = 0u;
    }
}

// ---------------- kernel 2: WMMA distance + argmin ----------------
// 64 threads = 2 waves; block handles 64 M-rows; each wave owns 32 rows
// (two 16-row WMMA tiles sharing every B fragment -> 2x arithmetic
// intensity vs one tile per wave).
__global__ void __launch_bounds__(64)
vq_argmin(const float* __restrict__ x, const __bf16* __restrict__ ebf,
          const float* __restrict__ enorm, unsigned* __restrict__ idx,
          unsigned* __restrict__ counts) {
    __shared__ __attribute__((aligned(16))) __bf16 lds_x[64 * 272]; // 272 = 256 + pad
    __shared__ float lds_en[NUM_EMB];

    const int tid = threadIdx.x;
    const int m0  = blockIdx.x * 64;
    const int b   = m0 >> 12;       // batch (64 rows never straddle a batch)
    const int mb  = m0 & 4095;      // offset inside the HW plane

    // ---- stage x tile (64 rows x 256 dims) to LDS as bf16 ----
    {
        const float* src = x + (size_t)b * EDIM * HW + mb + tid;  // row = tid
#pragma unroll 4
        for (int e = 0; e < EDIM; ++e)
            lds_x[tid * 272 + e] = (__bf16)src[(size_t)e * HW];
        for (int i = tid; i < NUM_EMB; i += 64) lds_en[i] = enorm[i];
    }
    __syncthreads();

    const int wv   = tid >> 5;      // wave 0..1
    const int ln   = tid & 31;
    const int half = ln >> 4;
    const int l16  = ln & 15;

    // ---- preload both A tiles into registers (16-bit A 16x32 layout) ----
    v16bf A0[8], A1[8];
    {
        const int r0 = (wv * 32 + l16) * 272;        // tile 0: rows wv*32 + 0..15
        const int r1 = r0 + 16 * 272;                // tile 1: rows wv*32 + 16..31
#pragma unroll
        for (int kc = 0; kc < 8; ++kc) {
            const int ks = kc * 32 + half * 16;
            v8bf a0lo = *(const v8bf*)&lds_x[r0 + ks];
            v8bf a0hi = *(const v8bf*)&lds_x[r0 + ks + 8];
            A0[kc] = __builtin_shufflevector(a0lo, a0hi,
                       0,1,2,3,4,5,6,7,8,9,10,11,12,13,14,15);
            v8bf a1lo = *(const v8bf*)&lds_x[r1 + ks];
            v8bf a1hi = *(const v8bf*)&lds_x[r1 + ks + 8];
            A1[kc] = __builtin_shufflevector(a1lo, a1hi,
                       0,1,2,3,4,5,6,7,8,9,10,11,12,13,14,15);
        }
    }

    float best0[8], best1[8];
    int   bidx0[8], bidx1[8];
#pragma unroll
    for (int r = 0; r < 8; ++r) {
        best0[r] = 3.4e38f; bidx0[r] = 0;
        best1[r] = 3.4e38f; bidx1[r] = 0;
    }

    for (int nt = 0; nt < 64; ++nt) {
        const int nabs = nt * 16 + l16;                     // B column for this lane
        const __bf16* brow = ebf + (size_t)nabs * EDIM + half * 16;
        if (nt < 63)   // pull next n-tile's codebook rows toward the WGP
            __builtin_prefetch(brow + 16 * EDIM, 0, 3);

        v8f acc0 = {}, acc1 = {};
#pragma unroll
        for (int kc = 0; kc < 8; ++kc) {
            // B fragment: 16 contiguous K values per lane (k = 16*half + s), same order as A
            v16bf bm = *(const v16bf*)(brow + kc * 32);
            acc0 = __builtin_amdgcn_wmma_f32_16x16x32_bf16(
                       false, A0[kc], false, bm, (short)0, acc0, false, false);
            acc1 = __builtin_amdgcn_wmma_f32_16x16x32_bf16(
                       false, A1[kc], false, bm, (short)0, acc1, false, false);
        }
        // score = ||e||^2 - 2 x.e   (||x||^2 constant per row -> dropped)
        const float en = lds_en[nabs];
#pragma unroll
        for (int r = 0; r < 8; ++r) {
            const float s0 = en - 2.0f * acc0[r];
            if (s0 < best0[r]) { best0[r] = s0; bidx0[r] = nabs; }
            const float s1 = en - 2.0f * acc1[r];
            if (s1 < best1[r]) { best1[r] = s1; bidx1[r] = nabs; }
        }
    }

    // reduce across the 16 lanes sharing each M row (C layout: N = lane%16)
#pragma unroll
    for (int r = 0; r < 8; ++r) {
#pragma unroll
        for (int off = 8; off; off >>= 1) {
            float ov = __shfl_xor(best0[r], off, 32);
            int   oi = __shfl_xor(bidx0[r], off, 32);
            if (ov < best0[r] || (ov == best0[r] && oi < bidx0[r])) {
                best0[r] = ov; bidx0[r] = oi;
            }
            ov = __shfl_xor(best1[r], off, 32);
            oi = __shfl_xor(bidx1[r], off, 32);
            if (ov < best1[r] || (ov == best1[r] && oi < bidx1[r])) {
                best1[r] = ov; bidx1[r] = oi;
            }
        }
    }
    if (l16 == 0) {   // lanes 0 (M rows +0..7) and 16 (M rows +8..15) per tile
#pragma unroll
        for (int r = 0; r < 8; ++r) {
            const int m = m0 + wv * 32 + half * 8 + r;
            idx[m] = (unsigned)bidx0[r];
            atomicAdd(&counts[bidx0[r]], 1u);
            idx[m + 16] = (unsigned)bidx1[r];
            atomicAdd(&counts[bidx1[r]], 1u);
        }
    }
}

// ---------------- kernel 3: gather codes into st & quantized ----------------
__global__ void vq_gather(const float* __restrict__ emb, const unsigned* __restrict__ idx,
                          float* __restrict__ out) {
    const int tid = threadIdx.x;
    const int m0  = blockIdx.x * 64;
    const int ml  = tid & 63;
    const int e0  = tid >> 6;      // 0..3
    const int m   = m0 + ml;
    const int b   = m >> 12;
    const int hw  = m & 4095;
    const unsigned k = idx[m];
    const float* crow = emb + (size_t)k * EDIM;
    float* st = out + (size_t)b * EDIM * HW + hw;
#pragma unroll 4
    for (int c = 0; c < 64; ++c) {
        const int e = 4 * c + e0;
        const float v = crow[e];
        st[(size_t)e * HW]             = v;   // st (== quantized numerically)
        st[(size_t)e * HW + OUT_PLANE] = v;   // quantized
    }
}

// ---------------- kernel 4: perplexity ----------------
__global__ void vq_perp(const unsigned* __restrict__ counts, float* __restrict__ out) {
    __shared__ float sp[32];
    __shared__ float stot;
    const int t = threadIdx.x;                 // 0..1023
    const float c = (float)counts[t];

    float v = c;
#pragma unroll
    for (int off = 16; off; off >>= 1) v += __shfl_xor(v, off, 32);
    if ((t & 31) == 0) sp[t >> 5] = v;
    __syncthreads();
    if (t < 32) {
        float s = sp[t];
#pragma unroll
        for (int off = 16; off; off >>= 1) s += __shfl_xor(s, off, 32);
        if (t == 0) stot = s;
    }
    __syncthreads();

    const float p = c / (stot + 1e-10f);
    float e = (p > 0.f) ? -p * logf(p) : 0.f;
#pragma unroll
    for (int off = 16; off; off >>= 1) e += __shfl_xor(e, off, 32);
    if ((t & 31) == 0) sp[t >> 5] = e;
    __syncthreads();
    if (t == 0) {
        float s = 0.f;
#pragma unroll
        for (int i = 0; i < 32; ++i) s += sp[i];
        out[2 * (size_t)OUT_PLANE] = expf(s);
    }
}

extern "C" void kernel_launch(void* const* d_in, const int* in_sizes, int n_in,
                              void* d_out, int out_size, void* d_ws, size_t ws_size,
                              hipStream_t stream) {
    const float* x   = (const float*)d_in[0];   // (16,256,64,64) fp32
    const float* emb = (const float*)d_in[1];   // (1024,256) fp32
    float* out = (float*)d_out;                 // st | quantized | perplexity

    char* ws = (char*)d_ws;
    __bf16*   ebf    = (__bf16*)ws;                    // 512 KB
    float*    enorm  = (float*)(ws + 524288);          // 4 KB
    unsigned* counts = (unsigned*)(ws + 528384);       // 4 KB
    unsigned* idxbuf = (unsigned*)(ws + 532480);       // 256 KB

    vq_prep  <<<NUM_EMB,      256, 0, stream>>>(emb, ebf, enorm, counts);
    vq_argmin<<<M_TOTAL / 64,  64, 0, stream>>>(x, ebf, enorm, idxbuf, counts);
    vq_gather<<<M_TOTAL / 64, 256, 0, stream>>>(emb, idxbuf, out);
    vq_perp  <<<1,           1024, 0, stream>>>(counts, out);
}
